// Block_33097017983119
// MI455X (gfx1250) — matrix operand order repaired and checked by reading, not compiled, not run on hardware
//
#include <hip/hip_runtime.h>

// ---------------------------------------------------------------------------
// MI455X (gfx1250) transformer block forward.
// Compute-bound (~137 GFLOP vs ~200MB traffic @ 23.3 TB/s) -> all GEMM-like
// work routed through v_wmma_f32_16x16x32_bf16 (f32 accumulate), with
// double-buffered GLOBAL_LOAD_ASYNC_TO_LDS_B128 (ASYNCcnt) feeding the GEMMs.
// Shapes: B=2, T=2048, D=1024, H=16, hd=64.  All GEMM dims multiples of 128.
// ---------------------------------------------------------------------------

typedef __bf16 bf16;
typedef __attribute__((ext_vector_type(8)))  __bf16 v8bf;
typedef __attribute__((ext_vector_type(16))) __bf16 v16bf;
typedef __attribute__((ext_vector_type(8)))  float  v8f;

__device__ __forceinline__ bf16 f2bf(float f) {
  unsigned u = __builtin_bit_cast(unsigned, f);
  u += 0x7FFFu + ((u >> 16) & 1u);           // round to nearest even
  unsigned short h = (unsigned short)(u >> 16);
  return __builtin_bit_cast(bf16, h);
}

__device__ __forceinline__ v16bf cat8(v8bf lo, v8bf hi) {
  return __builtin_shufflevector(lo, hi, 0,1,2,3,4,5,6,7,8,9,10,11,12,13,14,15);
}

__device__ __forceinline__ v8f vzero8() {
  v8f v;
#pragma unroll
  for (int i = 0; i < 8; ++i) v[i] = 0.f;
  return v;
}

// Async global -> LDS copy, 16B per lane, tracked by ASYNCcnt (CDNA5 TDM-lite
// path, CDNA5 ISA).  lds_off = wave-relative LDS byte
// address (low 32 bits of the generic shared pointer), ga = per-lane global
// address (GV mode).
__device__ __forceinline__ void async_ld_b128(unsigned lds_off, const void* ga) {
  asm volatile("global_load_async_to_lds_b128 %0, %1, off"
               :: "v"(lds_off), "v"((unsigned long long)(uintptr_t)ga)
               : "memory");
}
__device__ __forceinline__ void wait_async0() {
  asm volatile("s_wait_asynccnt 0" ::: "memory");
}

// ---------------------------------------------------------------------------
// Tiled f32 -> bf16 transpose: dst[c*R + r] = bf16(src[r*C + c]).
// batch over blockIdx.z (per-head weight slabs). block = (32,8).
// ---------------------------------------------------------------------------
__global__ __launch_bounds__(256) void k_transpose_bf16(
    const float* __restrict__ src, bf16* __restrict__ dst, int R, int C)
{
  __shared__ float tile[32][33];
  const size_t bofs = (size_t)blockIdx.z * R * C;
  src += bofs; dst += bofs;
  const int r0 = blockIdx.y * 32, c0 = blockIdx.x * 32;
  const int tx = threadIdx.x, ty = threadIdx.y;
#pragma unroll
  for (int i = 0; i < 32; i += 8) {
    int r = r0 + ty + i, c = c0 + tx;
    tile[ty + i][tx] = (r < R && c < C) ? src[(size_t)r * C + c] : 0.f;
  }
  __syncthreads();
#pragma unroll
  for (int i = 0; i < 32; i += 8) {
    int c = c0 + ty + i, r = r0 + tx;
    if (c < C && r < R) dst[(size_t)c * R + r] = f2bf(tile[tx][ty + i]);
  }
}

// ---------------------------------------------------------------------------
// LayerNorm, one wave32 per row of D=1024 f32, bf16 output.
// ---------------------------------------------------------------------------
template <int D>
__global__ __launch_bounds__(256) void k_layernorm_bf16(
    const float* __restrict__ x, const float* __restrict__ g,
    const float* __restrict__ beta, bf16* __restrict__ out)
{
  const int wave = threadIdx.x >> 5, lane = threadIdx.x & 31;
  const int row  = blockIdx.x * 8 + wave;
  const float* xr = x + (size_t)row * D;
  constexpr int N = D / 32;
  float vals[N];
  float s = 0.f, ss = 0.f;
#pragma unroll
  for (int i = 0; i < N; ++i) {
    float v = xr[i * 32 + lane];
    vals[i] = v; s += v; ss += v * v;
  }
#pragma unroll
  for (int m = 16; m >= 1; m >>= 1) {
    s  += __shfl_xor(s,  m, 32);
    ss += __shfl_xor(ss, m, 32);
  }
  const float mu   = s / (float)D;
  const float var  = ss / (float)D - mu * mu;
  const float rstd = rsqrtf(var + 1e-5f);
  bf16* orow = out + (size_t)row * D;
#pragma unroll
  for (int i = 0; i < N; ++i) {
    int c = i * 32 + lane;
    orow[c] = f2bf((vals[i] - mu) * rstd * g[c] + beta[c]);
  }
}

// ---------------------------------------------------------------------------
// bf16 WMMA GEMM: C[M,N] = A[M,K] * Bt[N,K]^T  (Bt = B transposed, row-major).
// Block tile 128x128, 8 waves, each wave 32x64 (2x4 16x16 C tiles), K-step 32.
// Double-buffered async global->LDS staging (ASYNCcnt).
// MODE 0: bf16 out.  MODE 1: f32 out + bias + residual.  MODE 2: bf16 relu(acc+bias).
// Requires M%128==0, N%128==0, K%32==0 (true for all uses here).
// ---------------------------------------------------------------------------
template <int MODE>
__global__ __launch_bounds__(256) void k_gemm_bf16(
    const bf16* __restrict__ A, int lda,
    const bf16* __restrict__ Bt, int ldb,
    void* __restrict__ Out, int ldo,
    const float* __restrict__ bias,
    const float* __restrict__ resid,
    int K)
{
  __shared__ alignas(16) bf16 As[2][128 * 40];   // +8 pad: 80B stride, conflict-free
  __shared__ alignas(16) bf16 Bs[2][128 * 40];
  const int tid  = threadIdx.x;
  const int wave = tid >> 5, lane = tid & 31;
  const int hl   = lane >> 4, r = lane & 15;     // lane-half, lane-in-half
  const int wm   = (wave & 3) * 32, wn = (wave >> 2) * 64;
  const int bm   = blockIdx.y * 128, bn = blockIdx.x * 128;
  const int srow = tid >> 2, scol = (tid & 3) * 8;   // 64 rows/pass, 2 passes

  v8f acc[2][4];
#pragma unroll
  for (int i = 0; i < 2; ++i)
#pragma unroll
    for (int t = 0; t < 4; ++t) acc[i][t] = vzero8();

  // stage one 128x32 A tile + one 128x32 B tile into LDS buffer `buf`
  auto issue = [&](int buf, int k0) {
    const bf16* ga = &A [(size_t)(bm + srow) * lda + k0 + scol];
    const bf16* gb = &Bt[(size_t)(bn + srow) * ldb + k0 + scol];
    async_ld_b128((unsigned)(uintptr_t)&As[buf][srow * 40 + scol],        ga);
    async_ld_b128((unsigned)(uintptr_t)&As[buf][(srow + 64) * 40 + scol], ga + (size_t)64 * lda);
    async_ld_b128((unsigned)(uintptr_t)&Bs[buf][srow * 40 + scol],        gb);
    async_ld_b128((unsigned)(uintptr_t)&Bs[buf][(srow + 64) * 40 + scol], gb + (size_t)64 * ldb);
  };

  const int nk = K >> 5;
  issue(0, 0);
  for (int it = 0; it < nk; ++it) {
    wait_async0();                 // this wave's 4 async loads for buf landed
    __syncthreads();               // ... and everyone else's
    if (it + 1 < nk) issue((it + 1) & 1, (it + 1) << 5);   // prefetch next
    const bf16* as = &As[it & 1][0];
    const bf16* bs = &Bs[it & 1][0];

    v16bf a[2], b[4];
#pragma unroll
    for (int i = 0; i < 2; ++i) {
      // A 16x32 fragment: half0 elems = K{0..7,16..23}, half1 = K{8..15,24..31}
      const bf16* p = &as[(wm + 16 * i + r) * 40 + hl * 8];
      a[i] = cat8(*(const v8bf*)p, *(const v8bf*)(p + 16));
    }
#pragma unroll
    for (int t = 0; t < 4; ++t) {
      // B 32x16 fragment: half0 elems = K{0..7,16..23}, half1 = K{8..15,24..31} (lane = column)
      const bf16* p = &bs[(wn + 16 * t + r) * 40 + hl * 8];
      b[t] = cat8(*(const v8bf*)p, *(const v8bf*)(p + 16));
    }
#pragma unroll
    for (int i = 0; i < 2; ++i)
#pragma unroll
      for (int t = 0; t < 4; ++t)
        acc[i][t] = __builtin_amdgcn_wmma_f32_16x16x32_bf16(
            false, a[i], false, b[t], (short)0, acc[i][t], false, false);
    __syncthreads();               // all waves done reading buf before reissue
  }

  // C layout: reg j -> row (j + 8*half), lane%16 -> column
#pragma unroll
  for (int i = 0; i < 2; ++i)
#pragma unroll
    for (int t = 0; t < 4; ++t)
#pragma unroll
      for (int j = 0; j < 8; ++j) {
        const int gr = bm + wm + 16 * i + 8 * hl + j;
        const int gc = bn + wn + 16 * t + r;
        const float v = acc[i][t][j];
        if (MODE == 0) {
          ((bf16*)Out)[(size_t)gr * ldo + gc] = f2bf(v);
        } else if (MODE == 1) {
          ((float*)Out)[(size_t)gr * ldo + gc] =
              v + bias[gc] + resid[(size_t)gr * ldo + gc];
        } else {
          float u = v + bias[gc];
          ((bf16*)Out)[(size_t)gr * ldo + gc] = f2bf(u > 0.f ? u : 0.f);
        }
      }
}

// ---------------------------------------------------------------------------
// Causal flash attention. qkv: bf16 [B*T][3072] (q|k|v, head h at cols h*64).
// Grid: (T/64, B*H). 4 waves/WG: wave w owns q rows [qt0+16w, +16), all waves
// share staged K/V tiles (32 keys). hd=64, scale=1/8 applied to S.
// Output o: bf16 [B*T][1024] (heads concatenated).
// ---------------------------------------------------------------------------
__global__ __launch_bounds__(128) void k_attention(
    const bf16* __restrict__ qkv, bf16* __restrict__ o, int T, int H)
{
  __shared__ alignas(16) bf16 Ks[32 * 72];      // [key][hd], +8 pad
  __shared__ alignas(16) bf16 Vs[64 * 40];      // [hd][key] (transposed), +8 pad
  __shared__ alignas(16) bf16 Ps[4][16 * 40];   // per-wave P tile [q][key]
  const int tid = threadIdx.x, wave = tid >> 5, lane = tid & 31;
  const int hl = lane >> 4, r = lane & 15;
  const int bh = blockIdx.y, bb = bh / H, h = bh % H;
  const int qt0 = blockIdx.x * 64;
  const int qrow = qt0 + 16 * wave;
  const size_t rowbase = (size_t)(bb * T) * 3072 + h * 64;

  // Q fragments (A operand), loaded once: 16 rows x 64 (two K=32 chunks)
  v16bf aq[2];
  {
    const bf16* qp = qkv + rowbase + (size_t)(qrow + r) * 3072;
#pragma unroll
    for (int c = 0; c < 2; ++c) {
      const bf16* p = qp + 32 * c + hl * 8;
      aq[c] = cat8(*(const v8bf*)p, *(const v8bf*)(p + 16));
    }
  }

  v8f oacc[4];
#pragma unroll
  for (int t2 = 0; t2 < 4; ++t2) oacc[t2] = vzero8();
  float mrow[8], lrow[8];
#pragma unroll
  for (int j = 0; j < 8; ++j) { mrow[j] = -1e30f; lrow[j] = 0.f; }

  const int q_hi = qrow + 15;
  const int kend = qt0 + 64;                    // max keys needed by wave 3
  for (int k0 = 0; k0 < kend; k0 += 32) {
    // cooperative stage: K straight, V transposed into LDS
    for (int c = tid; c < 256; c += 128) {
      const int kr = c >> 3, c8 = (c & 7) * 8;
      const bf16* kp = qkv + (size_t)(bb * T + k0 + kr) * 3072 + 1024 + h * 64 + c8;
      *(v8bf*)&Ks[kr * 72 + c8] = *(const v8bf*)kp;
      v8bf vv = *(const v8bf*)(kp + 1024);      // v row
#pragma unroll
      for (int e = 0; e < 8; ++e) Vs[(c8 + e) * 40 + kr] = vv[e];
    }
    __syncthreads();
    if (k0 <= q_hi) {                           // wave-uniform: EXEC stays all-1s
      // S = Q K^T : two 16x16 key tiles, hd accumulated over 2 WMMAs each
      v8f s[2]; s[0] = vzero8(); s[1] = vzero8();
#pragma unroll
      for (int t = 0; t < 2; ++t)
#pragma unroll
        for (int c = 0; c < 2; ++c) {
          const bf16* p = &Ks[(16 * t + r) * 72 + 32 * c + hl * 8];
          v16bf bk = cat8(*(const v8bf*)p, *(const v8bf*)(p + 16));
          s[t] = __builtin_amdgcn_wmma_f32_16x16x32_bf16(
              false, aq[c], false, bk, (short)0, s[t], false, false);
        }
      // scale, causal mask, online softmax (row = j + 8*hl, col = lane%16)
      float pr[2][8];
#pragma unroll
      for (int j = 0; j < 8; ++j) {
        const int gq = qrow + 8 * hl + j;
        float rm = -1e30f;
#pragma unroll
        for (int t = 0; t < 2; ++t) {
          float v = s[t][j] * 0.125f;           // 1/sqrt(64)
          const int gk = k0 + 16 * t + r;
          v = (gk <= gq) ? v : -1e30f;
          pr[t][j] = v;
          rm = fmaxf(rm, v);
        }
        rm = fmaxf(rm, __shfl_xor(rm, 1, 32));
        rm = fmaxf(rm, __shfl_xor(rm, 2, 32));
        rm = fmaxf(rm, __shfl_xor(rm, 4, 32));
        rm = fmaxf(rm, __shfl_xor(rm, 8, 32));
        const float mnew  = fmaxf(mrow[j], rm);
        const float alpha = __expf(mrow[j] - mnew);
        float rs = 0.f;
#pragma unroll
        for (int t = 0; t < 2; ++t) {
          float pv = __expf(pr[t][j] - mnew);
          pr[t][j] = pv; rs += pv;
        }
        rs += __shfl_xor(rs, 1, 32);
        rs += __shfl_xor(rs, 2, 32);
        rs += __shfl_xor(rs, 4, 32);
        rs += __shfl_xor(rs, 8, 32);
        lrow[j] = lrow[j] * alpha + rs;
        mrow[j] = mnew;
#pragma unroll
        for (int t2 = 0; t2 < 4; ++t2) oacc[t2][j] *= alpha;
      }
      // re-shape P: C layout -> A fragment via wave-private LDS (in-order DS)
      bf16* pw = &Ps[wave][0];
#pragma unroll
      for (int t = 0; t < 2; ++t)
#pragma unroll
        for (int j = 0; j < 8; ++j)
          pw[(8 * hl + j) * 40 + 16 * t + r] = f2bf(pr[t][j]);
      v16bf ap;
      {
        const bf16* p = pw + r * 40 + hl * 8;
        ap = cat8(*(const v8bf*)p, *(const v8bf*)(p + 16));
      }
      // O += P V : 4 output 16x16 tiles over hd
#pragma unroll
      for (int t2 = 0; t2 < 4; ++t2) {
        const bf16* p = &Vs[(16 * t2 + r) * 40 + 8 * hl];
        v16bf bv = cat8(*(const v8bf*)p, *(const v8bf*)(p + 16));
        oacc[t2] = __builtin_amdgcn_wmma_f32_16x16x32_bf16(
            false, ap, false, bv, (short)0, oacc[t2], false, false);
      }
    }
    __syncthreads();
  }
  // normalize and store (heads concatenated)
  bf16* ob = o + (size_t)(bb * T) * 1024 + h * 64;
#pragma unroll
  for (int t2 = 0; t2 < 4; ++t2)
#pragma unroll
    for (int j = 0; j < 8; ++j) {
      const int gq = qrow + 8 * hl + j;
      ob[(size_t)gq * 1024 + 16 * t2 + r] = f2bf(oacc[t2][j] / lrow[j]);
    }
}

// ---------------------------------------------------------------------------
extern "C" void kernel_launch(void* const* d_in, const int* in_sizes, int n_in,
                              void* d_out, int out_size, void* d_ws, size_t ws_size,
                              hipStream_t stream) {
  (void)in_sizes; (void)n_in; (void)out_size; (void)ws_size;
  const float* x     = (const float*)d_in[0];
  const float* wq    = (const float*)d_in[1];
  const float* wk    = (const float*)d_in[2];
  const float* wv    = (const float*)d_in[3];
  const float* wproj = (const float*)d_in[4];
  const float* bproj = (const float*)d_in[5];
  const float* wff1  = (const float*)d_in[6];
  const float* bff1  = (const float*)d_in[7];
  const float* wff2  = (const float*)d_in[8];
  const float* bff2  = (const float*)d_in[9];
  const float* ln1g  = (const float*)d_in[10];
  const float* ln1b  = (const float*)d_in[11];
  const float* ln2g  = (const float*)d_in[12];
  const float* ln2b  = (const float*)d_in[13];

  const int B = 2, T = 2048, D = 1024, H = 16, hd = 64;
  const int M = B * T;                      // 4096 rows

  // workspace carve-out (~126 MB total)
  char* ws = (char*)d_ws;
  size_t off = 0;
  auto alloc = [&](size_t bytes) -> char* {
    char* p = ws + off;
    off += (bytes + 255) & ~(size_t)255;
    return p;
  };
  bf16*  qkvT   = (bf16*)alloc((size_t)3 * D * D * 2);   // [3072][1024] Bt for QKV
  bf16*  wprojT = (bf16*)alloc((size_t)D * D * 2);
  bf16*  wff1T  = (bf16*)alloc((size_t)4 * D * D * 2);
  bf16*  wff2T  = (bf16*)alloc((size_t)4 * D * D * 2);
  bf16*  hb     = (bf16*)alloc((size_t)M * D * 2);       // LN1 out
  bf16*  qkv    = (bf16*)alloc((size_t)M * 3 * D * 2);   // [4096][3072]
  bf16*  ob     = (bf16*)alloc((size_t)M * D * 2);       // attention out
  float* x1     = (float*)alloc((size_t)M * D * 4);      // x + attn proj
  bf16*  h2b    = (bf16*)alloc((size_t)M * D * 2);       // LN2 out
  bf16*  ff     = (bf16*)alloc((size_t)M * 4 * D * 2);   // relu(ff1)

  const dim3 tb(32, 8);
  // weights -> bf16, transposed to Bt[N][K]; per-head slabs batched on z
  k_transpose_bf16<<<dim3(hd/32, D/32, H), tb, 0, stream>>>(wq, qkvT,           D, hd);
  k_transpose_bf16<<<dim3(hd/32, D/32, H), tb, 0, stream>>>(wk, qkvT +   D * D, D, hd);
  k_transpose_bf16<<<dim3(hd/32, D/32, H), tb, 0, stream>>>(wv, qkvT + 2*D * D, D, hd);
  k_transpose_bf16<<<dim3(D/32,   D/32, 1), tb, 0, stream>>>(wproj, wprojT, D, D);
  k_transpose_bf16<<<dim3(4*D/32, D/32, 1), tb, 0, stream>>>(wff1,  wff1T,  D, 4*D);
  k_transpose_bf16<<<dim3(D/32, 4*D/32, 1), tb, 0, stream>>>(wff2,  wff2T,  4*D, D);

  // --- attention sublayer ---
  k_layernorm_bf16<1024><<<M / 8, 256, 0, stream>>>(x, ln1g, ln1b, hb);
  k_gemm_bf16<0><<<dim3(3*D/128, M/128), 256, 0, stream>>>(
      hb, D, qkvT, D, qkv, 3*D, nullptr, nullptr, D);
  k_attention<<<dim3(T/64, B*H), 128, 0, stream>>>(qkv, ob, T, H);
  k_gemm_bf16<1><<<dim3(D/128, M/128), 256, 0, stream>>>(
      ob, D, wprojT, D, x1, D, bproj, x, D);

  // --- FFN sublayer ---
  k_layernorm_bf16<1024><<<M / 8, 256, 0, stream>>>(x1, ln2g, ln2b, h2b);
  k_gemm_bf16<2><<<dim3(4*D/128, M/128), 256, 0, stream>>>(
      h2b, D, wff1T, D, ff, 4*D, bff1, nullptr, D);
  k_gemm_bf16<1><<<dim3(D/128, M/128), 256, 0, stream>>>(
      ff, 4*D, wff2T, 4*D, (float*)d_out, D, bff2, x1, 4*D);
}
